// TriangleMultiplicativeUpdate_23596550324671
// MI455X (gfx1250) — matrix operand order repaired and checked by reading, not compiled, not run on hardware
//
#include <hip/hip_runtime.h>
#include <hip/hip_bf16.h>

typedef __attribute__((ext_vector_type(16))) _Float16 v16h;
typedef __attribute__((ext_vector_type(8)))  _Float16 v8h;
typedef __attribute__((ext_vector_type(8)))  float    v8f;
typedef __attribute__((ext_vector_type(4)))  float    v4f;

#define NSEQ 512
#define NN   (NSEQ * NSEQ)   // 262144 rows (i,j)
#define CZ   128

#if __has_builtin(__builtin_amdgcn_global_load_async_to_lds_b128)
#define HAS_ASYNC_LDS 1
typedef int v4i_ __attribute__((__vector_size__(16)));
typedef __attribute__((address_space(1))) v4i_ gv4i_t;  // global int4
typedef __attribute__((address_space(3))) v4i_ lv4i_t;  // LDS int4
#else
#define HAS_ASYNC_LDS 0
#endif

#if __has_builtin(__builtin_amdgcn_s_wait_asynccnt)
#define WAIT_ASYNCCNT(n) __builtin_amdgcn_s_wait_asynccnt(n)
#else
#define WAIT_ASYNCCNT(n) asm volatile("s_wait_asynccnt %0" ::"n"(n) : "memory")
#endif

// ---------------- WMMA helpers (CDNA5 16x16x32 f16, f32 accum) ----------------

__device__ __forceinline__ v8f wmma_f16(v16h a, v16h b, v8f c) {
  // (neg_a, A, neg_b, B, c_mod, C, reuse_a, reuse_b)
  return __builtin_amdgcn_wmma_f32_16x16x32_f16(false, a, false, b, (short)0, c,
                                                false, false);
}

__device__ __forceinline__ v8f zero8() {
  v8f r;
#pragma unroll
  for (int i = 0; i < 8; ++i) r[i] = 0.0f;
  return r;
}

// A-fragment 16x32 f16: rows row0..row0+15 of row-major src (leading dim ld).
// ISA layout: lanes 0-15 hold K = 0..7 (V0..3) and 16..23 (V4..7) of row (lane&15);
// lanes 16-31 hold K = 8..15 and 24..31.  -> two contiguous 16B runs per lane.
__device__ __forceinline__ v16h load_a_frag(const _Float16* src, int ld,
                                            int row0, int k0, int lane) {
  const _Float16* p =
      src + (size_t)(row0 + (lane & 15)) * ld + k0 + ((lane >> 4) << 3);
  v8h lo = *(const v8h*)(p);
  v8h hi = *(const v8h*)(p + 16);
  return __builtin_shufflevector(lo, hi, 0, 1, 2, 3, 4, 5, 6, 7, 8, 9, 10, 11,
                                 12, 13, 14, 15);
}

// B-fragment 32x16 f16 where element (k,n) = srcT[col0+n][k0+k] (srcT row-major,
// leading dim ld). ISA layout: lane holds column (lane&15), K = (lane>>4)*8 + 0..7
// and 16 + (lane>>4)*8 + 0..7 -> two contiguous 16B runs per lane.
__device__ __forceinline__ v16h load_b_frag(const _Float16* srcT, int ld,
                                            int col0, int k0, int lane) {
  const _Float16* p = srcT + (size_t)(col0 + (lane & 15)) * ld + k0 +
                      ((lane >> 4) << 3);
  v8h lo = *(const v8h*)(p);
  v8h hi = *(const v8h*)(p + 16);
  return __builtin_shufflevector(lo, hi, 0, 1, 2, 3, 4, 5, 6, 7, 8, 9, 10, 11,
                                 12, 13, 14, 15);
}

// LDS variants for the 32-wide staged panels (leading dim 40 halfwords = 80B,
// 16B-aligned rows, conflict-free bank stride).
__device__ __forceinline__ v16h lds_a_frag(const _Float16* s, int row0,
                                           int lane) {
  const _Float16* p = s + (row0 + (lane & 15)) * 40 + ((lane >> 4) << 3);
  v8h lo = *(const v8h*)(p);
  v8h hi = *(const v8h*)(p + 16);
  return __builtin_shufflevector(lo, hi, 0, 1, 2, 3, 4, 5, 6, 7, 8, 9, 10, 11,
                                 12, 13, 14, 15);
}

__device__ __forceinline__ v16h lds_b_frag(const _Float16* s, int col0,
                                           int lane) {
  const _Float16* p = s + (col0 + (lane & 15)) * 40 + ((lane >> 4) << 3);
  v8h lo = *(const v8h*)(p);
  v8h hi = *(const v8h*)(p + 16);
  return __builtin_shufflevector(lo, hi, 0, 1, 2, 3, 4, 5, 6, 7, 8, 9, 10, 11,
                                 12, 13, 14, 15);
}

__device__ __forceinline__ float sigmoidf_(float x) {
  return 1.0f / (1.0f + __expf(-x));
}

// ---------------- Kernel 0: cast+transpose weights to f16 [n][k] ----------------
__global__ __launch_bounds__(128) void tmu_prep_weights(
    const float* __restrict__ w_ap, const float* __restrict__ w_ag,
    const float* __restrict__ w_bp, const float* __restrict__ w_bg,
    const float* __restrict__ w_g, const float* __restrict__ w_z,
    _Float16* __restrict__ wt) {
  const int mat = blockIdx.x >> 7;
  const int n = blockIdx.x & 127;
  const float* W = (mat == 0)   ? w_ap
                   : (mat == 1) ? w_ag
                   : (mat == 2) ? w_bp
                   : (mat == 3) ? w_bg
                   : (mat == 4) ? w_g
                                : w_z;
  const int k = threadIdx.x;
  wt[((mat << 7) + n) * 128 + k] = (_Float16)W[k * 128 + n];
}

// ---------------- Kernel 1: LN(z) + 5 fused WMMA projections ----------------
__global__ __launch_bounds__(256) void tmu_proj_kernel(
    const float* __restrict__ z, const float* __restrict__ mask,
    const float* __restrict__ b_ap, const float* __restrict__ b_ag,
    const float* __restrict__ b_bp, const float* __restrict__ b_bg,
    const float* __restrict__ b_g, const float* __restrict__ ln_w,
    const float* __restrict__ ln_b, const _Float16* __restrict__ wt,
    _Float16* __restrict__ a_t,   // [CZ][NN] f16
    _Float16* __restrict__ b_t,   // [CZ][NN] f16
    float* __restrict__ gbuf) {   // [NN][CZ] f32
  __shared__ __align__(32) _Float16 zn[128][144];
  const int row0 = blockIdx.x * 128;
  const int tid = threadIdx.x;
  const int lane = tid & 31;
  const int wave = tid >> 5;

  // ---- Phase 1: LayerNorm (2 threads per row, shuffle-combined) ----
  {
    const int row = tid >> 1;
    const int half = tid & 1;
    const float* zr = z + (size_t)(row0 + row) * CZ + half * 64;
    float s = 0.f, s2 = 0.f;
#pragma unroll
    for (int i = 0; i < 64; i += 4) {
      v4f v = *(const v4f*)(zr + i);
#pragma unroll
      for (int j = 0; j < 4; ++j) {
        s += v[j];
        s2 += v[j] * v[j];
      }
    }
    s += __shfl_xor(s, 1);
    s2 += __shfl_xor(s2, 1);
    const float mu = s * (1.0f / 128.0f);
    const float var = s2 * (1.0f / 128.0f) - mu * mu;
    const float inv = rsqrtf(var + 1e-5f);
#pragma unroll
    for (int i = 0; i < 64; i += 4) {
      v4f v = *(const v4f*)(zr + i);
#pragma unroll
      for (int j = 0; j < 4; ++j) {
        const int ch = half * 64 + i + j;
        zn[row][ch] = (_Float16)((v[j] - mu) * inv * ln_w[ch] + ln_b[ch]);
      }
    }
  }
  __syncthreads();

  const _Float16* lds = &zn[0][0];
  const _Float16* wt_ap = wt;
  const _Float16* wt_ag = wt + 16384;
  const _Float16* wt_bp = wt + 2 * 16384;
  const _Float16* wt_bg = wt + 3 * 16384;
  const _Float16* wt_g = wt + 4 * 16384;

  // ---- Phase 2a: gated a/b projections (transposed: D[m=c][n=r]) ----
  for (int t = wave; t < 64; t += 8) {
    const int ct = (t >> 3) << 4;
    const int rt = (t & 7) << 4;
    v8f ap = zero8(), ag = zero8(), bp = zero8(), bg = zero8();
#pragma unroll
    for (int k0 = 0; k0 < 128; k0 += 32) {
      v16h bf = load_b_frag(lds, 144, rt, k0, lane);  // zn rows as B columns
      ap = wmma_f16(load_a_frag(wt_ap, 128, ct, k0, lane), bf, ap);
      ag = wmma_f16(load_a_frag(wt_ag, 128, ct, k0, lane), bf, ag);
      bp = wmma_f16(load_a_frag(wt_bp, 128, ct, k0, lane), bf, bp);
      bg = wmma_f16(load_a_frag(wt_bg, 128, ct, k0, lane), bf, bg);
    }
    const int rg = row0 + rt + (lane & 15);
    const float mk = mask[rg];
    const int msel = (lane >> 4) << 3;
#pragma unroll
    for (int v = 0; v < 8; ++v) {
      const int c = ct + v + msel;
      const float av = (ap[v] + b_ap[c]) * sigmoidf_(ag[v] + b_ag[c]) * mk;
      const float bv = (bp[v] + b_bp[c]) * sigmoidf_(bg[v] + b_bg[c]) * mk;
      a_t[(size_t)c * NN + rg] = (_Float16)av;
      b_t[(size_t)c * NN + rg] = (_Float16)bv;
    }
  }

  // ---- Phase 2b: output gate g (normal: D[m=r][n=c]) ----
  for (int t = wave; t < 64; t += 8) {
    const int rt = (t >> 3) << 4;
    const int ct = (t & 7) << 4;
    v8f acc = zero8();
#pragma unroll
    for (int k0 = 0; k0 < 128; k0 += 32)
      acc = wmma_f16(load_a_frag(lds, 144, rt, k0, lane),
                     load_b_frag(wt_g, 128, ct, k0, lane), acc);
    const int cl = ct + (lane & 15);
    const float bgc = b_g[cl];
    const int msel = (lane >> 4) << 3;
#pragma unroll
    for (int v = 0; v < 8; ++v) {
      const int rg = row0 + rt + v + msel;
      gbuf[(size_t)rg * CZ + cl] = sigmoidf_(acc[v] + bgc);
    }
  }
}

// ---------------- Kernel 2: triangle einsum, per-channel 512^3 GEMM ----------------
// X_c = A_c * B_c^T. a_t/b_t are [c][row][k] f16 (128 MB -> L2 resident).
// Block = 8 waves over a 64(i) x 128(j) tile; wave = 32x32 via 2x2 WMMA blocking.
// Async double-buffered LDS staging: per k-step, stage A 64x32 + B 128x32 panels
// with global_load_async_to_lds_b128 (3 ops/wave/batch), sync via s_wait_asynccnt.
__global__ __launch_bounds__(256) void tmu_triangle_kernel(
    const _Float16* __restrict__ a_t, const _Float16* __restrict__ b_t,
    float* __restrict__ x) {  // [i][j][c] f32
  const int c = blockIdx.z;
  const _Float16* A = a_t + (size_t)c * NN;  // [512][512]
  const _Float16* B = b_t + (size_t)c * NN;
  const int tid = threadIdx.x;
  const int lane = tid & 31;
  const int wave = tid >> 5;
  const int iblk = blockIdx.y * 64;
  const int jblk = blockIdx.x * 128;
  const int ai0 = (wave >> 2) * 32;  // local A rows for this wave
  const int bj0 = (wave & 3) * 32;   // local B rows (j columns) for this wave
  const int i0 = iblk + ai0;
  const int j0 = jblk + bj0;

  v8f acc[2][2] = {{zero8(), zero8()}, {zero8(), zero8()}};

#if HAS_ASYNC_LDS
  __shared__ __align__(16) _Float16 As[2][64][40];
  __shared__ __align__(16) _Float16 Bs[2][128][40];

  auto issue_batch = [&](int kk, int buf) {
    {  // A panel: 64 rows x 32 k = 256 x 16B segments, one per thread
      const int row = tid >> 2, seg = tid & 3;
      const _Float16* src = A + (size_t)(iblk + row) * NSEQ + kk + seg * 8;
      _Float16* dst = &As[buf][row][seg * 8];
      __builtin_amdgcn_global_load_async_to_lds_b128((gv4i_t*)src,
                                                     (lv4i_t*)dst, 0, 0);
    }
#pragma unroll
    for (int it = 0; it < 2; ++it) {  // B panel: 128 rows x 32 k = 512 segments
      const int idx = tid + (it << 8);
      const int row = idx >> 2, seg = idx & 3;
      const _Float16* src = B + (size_t)(jblk + row) * NSEQ + kk + seg * 8;
      _Float16* dst = &Bs[buf][row][seg * 8];
      __builtin_amdgcn_global_load_async_to_lds_b128((gv4i_t*)src,
                                                     (lv4i_t*)dst, 0, 0);
    }
  };

  issue_batch(0, 0);
  for (int s = 0; s < 16; ++s) {
    if (s < 15) {
      issue_batch((s + 1) * 32, (s + 1) & 1);
      // 3 ops of the new batch may remain in flight; batch s has retired
      // (async loads complete in order per the ASYNCcnt rules).
      WAIT_ASYNCCNT(3);
    } else {
      WAIT_ASYNCCNT(0);
    }
    __syncthreads();
    const int cur = s & 1;
    const _Float16* As_c = &As[cur][0][0];
    const _Float16* Bs_c = &Bs[cur][0][0];
    v16h a0 = lds_a_frag(As_c, ai0, lane);
    v16h a1 = lds_a_frag(As_c, ai0 + 16, lane);
    v16h b0 = lds_b_frag(Bs_c, bj0, lane);
    v16h b1 = lds_b_frag(Bs_c, bj0 + 16, lane);
    acc[0][0] = wmma_f16(a0, b0, acc[0][0]);
    acc[0][1] = wmma_f16(a0, b1, acc[0][1]);
    acc[1][0] = wmma_f16(a1, b0, acc[1][0]);
    acc[1][1] = wmma_f16(a1, b1, acc[1][1]);
    __syncthreads();  // protect buffer cur before batch s+2 overwrites it
  }
#else
  for (int k0 = 0; k0 < NSEQ; k0 += 32) {
    v16h a0 = load_a_frag(A, NSEQ, i0, k0, lane);
    v16h a1 = load_a_frag(A, NSEQ, i0 + 16, k0, lane);
    v16h b0 = load_b_frag(B, NSEQ, j0, k0, lane);
    v16h b1 = load_b_frag(B, NSEQ, j0 + 16, k0, lane);
    acc[0][0] = wmma_f16(a0, b0, acc[0][0]);
    acc[0][1] = wmma_f16(a0, b1, acc[0][1]);
    acc[1][0] = wmma_f16(a1, b0, acc[1][0]);
    acc[1][1] = wmma_f16(a1, b1, acc[1][1]);
  }
#endif

  const int msel = (lane >> 4) << 3;
#pragma unroll
  for (int mi = 0; mi < 2; ++mi)
#pragma unroll
    for (int nj = 0; nj < 2; ++nj) {
      const int jb = j0 + nj * 16 + (lane & 15);
#pragma unroll
      for (int v = 0; v < 8; ++v) {
        const int i = i0 + mi * 16 + v + msel;
        x[((size_t)i * NSEQ + jb) * CZ + c] = acc[mi][nj][v];
      }
    }
}

// ---------------- Kernel 3: LN(x) @ w_z + b_z, times gate ----------------
__global__ __launch_bounds__(256) void tmu_out_kernel(
    const float* __restrict__ x, const float* __restrict__ gbuf,
    const _Float16* __restrict__ wt_z, const float* __restrict__ b_z,
    const float* __restrict__ ln_w, const float* __restrict__ ln_b,
    float* __restrict__ out) {
  __shared__ __align__(32) _Float16 xn[128][144];
  const int row0 = blockIdx.x * 128;
  const int tid = threadIdx.x;
  const int lane = tid & 31;
  const int wave = tid >> 5;

  {
    const int row = tid >> 1;
    const int half = tid & 1;
    const float* xr = x + (size_t)(row0 + row) * CZ + half * 64;
    float s = 0.f, s2 = 0.f;
#pragma unroll
    for (int i = 0; i < 64; i += 4) {
      v4f v = *(const v4f*)(xr + i);
#pragma unroll
      for (int j = 0; j < 4; ++j) {
        s += v[j];
        s2 += v[j] * v[j];
      }
    }
    s += __shfl_xor(s, 1);
    s2 += __shfl_xor(s2, 1);
    const float mu = s * (1.0f / 128.0f);
    const float var = s2 * (1.0f / 128.0f) - mu * mu;
    const float inv = rsqrtf(var + 1e-5f);
#pragma unroll
    for (int i = 0; i < 64; i += 4) {
      v4f v = *(const v4f*)(xr + i);
#pragma unroll
      for (int j = 0; j < 4; ++j) {
        const int ch = half * 64 + i + j;
        xn[row][ch] = (_Float16)((v[j] - mu) * inv * ln_w[ch] + ln_b[ch]);
      }
    }
  }
  __syncthreads();

  const _Float16* lds = &xn[0][0];
  for (int t = wave; t < 64; t += 8) {
    const int rt = (t >> 3) << 4;
    const int ct = (t & 7) << 4;
    v8f acc = zero8();
#pragma unroll
    for (int k0 = 0; k0 < 128; k0 += 32)
      acc = wmma_f16(load_a_frag(lds, 144, rt, k0, lane),
                     load_b_frag(wt_z, 128, ct, k0, lane), acc);
    const int cl = ct + (lane & 15);
    const float bz = b_z[cl];
    const int msel = (lane >> 4) << 3;
#pragma unroll
    for (int v = 0; v < 8; ++v) {
      const int rg = row0 + rt + v + msel;
      out[(size_t)rg * CZ + cl] = (acc[v] + bz) * gbuf[(size_t)rg * CZ + cl];
    }
  }
}

// ---------------- Host launcher ----------------

// Workspace layout (all offsets 256B aligned):
#define OFF_WT 0                                 // 6*128*128 f16  = 196608 B
#define OFF_AT ((size_t)196608)                  // 64 MB f16 [c][row]
#define OFF_BT (OFF_AT + (size_t)CZ * NN * 2)    // 64 MB f16
#define OFF_G  (OFF_BT + (size_t)CZ * NN * 2)    // 128 MB f32 [row][c]
#define OFF_X  (OFF_G + (size_t)NN * CZ * 4)     // 128 MB f32 [row][c]

extern "C" void kernel_launch(void* const* d_in, const int* in_sizes, int n_in,
                              void* d_out, int out_size, void* d_ws,
                              size_t ws_size, hipStream_t stream) {
  const float* z = (const float*)d_in[0];
  const float* mask = (const float*)d_in[1];
  const float* w_ap = (const float*)d_in[2];
  const float* b_ap = (const float*)d_in[3];
  const float* w_ag = (const float*)d_in[4];
  const float* b_ag = (const float*)d_in[5];
  const float* w_bp = (const float*)d_in[6];
  const float* b_bp = (const float*)d_in[7];
  const float* w_bg = (const float*)d_in[8];
  const float* b_bg = (const float*)d_in[9];
  const float* w_g = (const float*)d_in[10];
  const float* b_g = (const float*)d_in[11];
  const float* w_z = (const float*)d_in[12];
  const float* b_z = (const float*)d_in[13];
  const float* ln_in_w = (const float*)d_in[14];
  const float* ln_in_b = (const float*)d_in[15];
  const float* ln_out_w = (const float*)d_in[16];
  const float* ln_out_b = (const float*)d_in[17];

  char* ws = (char*)d_ws;
  _Float16* wt = (_Float16*)(ws + OFF_WT);
  _Float16* a_t = (_Float16*)(ws + OFF_AT);
  _Float16* b_t = (_Float16*)(ws + OFF_BT);
  float* gbuf = (float*)(ws + OFF_G);
  float* xbuf = (float*)(ws + OFF_X);
  float* out = (float*)d_out;

  tmu_prep_weights<<<dim3(6 * 128), 128, 0, stream>>>(w_ap, w_ag, w_bp, w_bg,
                                                      w_g, w_z, wt);
  tmu_proj_kernel<<<dim3(NN / 128), 256, 0, stream>>>(
      z, mask, b_ap, b_ag, b_bp, b_bg, b_g, ln_in_w, ln_in_b, wt, a_t, b_t,
      gbuf);
  tmu_triangle_kernel<<<dim3(NSEQ / 128, NSEQ / 64, CZ), 256, 0, stream>>>(
      a_t, b_t, xbuf);
  tmu_out_kernel<<<dim3(NN / 128), 256, 0, stream>>>(
      xbuf, gbuf, wt + 5 * 16384, b_z, ln_out_w, ln_out_b, out);
}
